// Actformer_30313879176060
// MI455X (gfx1250) — matrix operand order repaired and checked by reading, not compiled, not run on hardware
//
#include <hip/hip_runtime.h>

#define B_ 1024
#define S_ 2048
#define D_ 64
#define H_ 4

typedef __attribute__((ext_vector_type(16))) __bf16 v16bf;
typedef __attribute__((ext_vector_type(8)))  float  v8f;

__launch_bounds__(256, 1)
__global__ void actformer_fused(
    const float* __restrict__ query,
    const float* __restrict__ scr,
    const float* __restrict__ wq, const float* __restrict__ wk, const float* __restrict__ wv,
    const float* __restrict__ bq, const float* __restrict__ bk, const float* __restrict__ bv,
    const float* __restrict__ wo, const float* __restrict__ bo,
    const float* __restrict__ w_write, const float* __restrict__ b_write,
    const float* __restrict__ w_a1, const float* __restrict__ b_a1,
    const float* __restrict__ w_a2, const float* __restrict__ b_a2,
    float* __restrict__ out)
{
    __shared__ float  s_tmp[64];
    __shared__ float  s_q[64];
    __shared__ __align__(16) __bf16 s_rqb[16][D_];   // bf16 rq, rows 4..15 zero-padded
    __shared__ float  s_cb[H_];
    __shared__ __align__(16) float s_sc[H_][S_];     // scores -> unnormalized exp(p)
    __shared__ float  s_part[8][H_][D_];             // per-wave partial weighted sums
    __shared__ float  s_xw[H_][D_];
    __shared__ float  s_ctx[64];
    __shared__ float  s_val[64];
    __shared__ float  s_hid[128];
    __shared__ __align__(16) float s_wvo[64];
    __shared__ float  s_lg[S_];                      // logits -> unnormalized exp
    __shared__ float  s_red[256];
    __shared__ float  s_hmax[H_], s_hsum[H_];
    __shared__ float  s_asc[2];                      // [0]=logit max, [1]=1/sum

    const int tid  = threadIdx.x;
    const int lane = tid & 31;
    const int wave = tid >> 5;
    const int b    = blockIdx.x;
    const float* __restrict__ xb = scr + (size_t)b * S_ * D_;

    // Warm w_a2 (1 MB, shared across all 1024 blocks) into L2.
    __builtin_prefetch(w_a2 + (size_t)tid * 1024, 0, 1);

    // ---------------- Phase 0: q = query @ wq^T + bq; fold into rq_bf16 (16x64 padded), cb (4)
    if (tid < 64) s_tmp[tid] = query[(size_t)b * 64 + tid];
    __syncthreads();
    if (tid < 64) {
        float acc = bq[tid];
        #pragma unroll 8
        for (int e = 0; e < 64; ++e) acc += s_tmp[e] * wq[tid * 64 + e];
        s_q[tid] = acc;
    }
    // zero the padding rows of s_rqb (rows 4..15) while q is computed
    for (int idx = tid; idx < 12 * 64; idx += 256)
        s_rqb[4 + (idx >> 6)][idx & 63] = (__bf16)0.f;
    __syncthreads();
    {
        int h = tid >> 6, e = tid & 63;
        float acc = 0.f;
        #pragma unroll
        for (int j = 0; j < 16; ++j) acc += s_q[h * 16 + j] * wk[(h * 16 + j) * 64 + e];
        s_rqb[h][e] = (__bf16)acc;
        if (tid < H_) {
            float cb = 0.f;
            #pragma unroll
            for (int j = 0; j < 16; ++j) cb += s_q[tid * 16 + j] * bk[tid * 16 + j];
            s_cb[tid] = cb;
        }
    }
    __syncthreads();

    // ---------------- Phase 1: scores[h][s] = 0.25*(x[s]·rq[h] + cb[h]) via WMMA bf16
    {
        const int m   = lane & 15;
        const int kb  = (lane & 16) ? 16 : 0;  // A: K-halves per lane group
        const int n   = lane & 15;
        const int g16 = (lane & 16) ? 16 : 0;  // B: row group per lane group
        const int mb  = (lane & 16) ? 8 : 0;

        // Hoisted loop-invariant B fragments (contiguous bf16 -> ds_load_b128)
        v16bf bm0, bm1;
        #pragma unroll
        for (int i = 0; i < 16; ++i) {
            bm0[i] = s_rqb[n][g16 + i];
            bm1[i] = s_rqb[n][32 + g16 + i];
        }

        for (int t = wave; t < S_ / 16; t += 8) {
            const int s0 = t * 16;
            const float* row = xb + (size_t)(s0 + m) * D_;
            v8f c = {};
            #pragma unroll
            for (int chunk = 0; chunk < 2; ++chunk) {
                const int c0 = chunk * 32 + kb;
                float4 f0 = *(const float4*)(row + c0);
                float4 f1 = *(const float4*)(row + c0 + 4);
                float4 f2 = *(const float4*)(row + c0 + 8);
                float4 f3 = *(const float4*)(row + c0 + 12);
                v16bf a;
                a[0]=(__bf16)f0.x; a[1]=(__bf16)f0.y; a[2]=(__bf16)f0.z; a[3]=(__bf16)f0.w;
                a[4]=(__bf16)f1.x; a[5]=(__bf16)f1.y; a[6]=(__bf16)f1.z; a[7]=(__bf16)f1.w;
                a[8]=(__bf16)f2.x; a[9]=(__bf16)f2.y; a[10]=(__bf16)f2.z; a[11]=(__bf16)f2.w;
                a[12]=(__bf16)f3.x; a[13]=(__bf16)f3.y; a[14]=(__bf16)f3.z; a[15]=(__bf16)f3.w;
                c = __builtin_amdgcn_wmma_f32_16x16x32_bf16(false, a, false,
                                                            (chunk == 0) ? bm0 : bm1,
                                                            (short)0, c, false, false);
            }
            if (n < H_) {
                const float cbn = s_cb[n];
                #pragma unroll
                for (int j = 0; j < 8; ++j)
                    s_sc[n][s0 + mb + j] = 0.25f * (c[j] + cbn);
            }
        }
    }
    __syncthreads();

    // ---------------- Phase 1b: per-head softmax (store unnormalized exp, keep sums)
    {
        int h = tid >> 6, lt = tid & 63;
        float mx = -1e30f;
        for (int s = lt; s < S_; s += 64) mx = fmaxf(mx, s_sc[h][s]);
        s_red[tid] = mx;
        __syncthreads();
        if (lt == 0) {
            float m = -1e30f;
            for (int i = 0; i < 64; ++i) m = fmaxf(m, s_red[h * 64 + i]);
            s_hmax[h] = m;
        }
        __syncthreads();
        float m = s_hmax[h], sum = 0.f;
        for (int s = lt; s < S_; s += 64) {
            float e = __expf(s_sc[h][s] - m);
            s_sc[h][s] = e;
            sum += e;
        }
        s_red[tid] = sum;
        __syncthreads();
        if (lt == 0) {
            float t = 0.f;
            for (int i = 0; i < 64; ++i) t += s_red[h * 64 + i];
            s_hsum[h] = t;
        }
    }
    __syncthreads();

    // ---------------- Phase 2: xw[h] = (Σ_s p[h,s]·x[s]) / Σp via WMMA (A=p 16x32, B=x 32x16)
    {
        const int m    = lane & 15;
        const int mm   = m & 3;                 // in-range row (speculatable LDS load)
        const bool mok = (m < H_);
        const int kb   = (lane & 16) ? 16 : 0;
        const int n    = lane & 15;
        const int g16  = (lane & 16) ? 16 : 0;
        v8f cacc[4] = {};
        for (int t = wave; t < S_ / 32; t += 8) {
            const int s0 = t * 32;
            v16bf a;
            #pragma unroll
            for (int i = 0; i < 16; ++i) {
                int k = kb + i;
                float v = s_sc[mm][s0 + k];     // unconditional, contiguous -> ds_load_b128
                a[i] = (__bf16)(mok ? v : 0.f); // value select, no control flow
            }
            #pragma unroll
            for (int g = 0; g < 4; ++g) {
                v16bf bm;
                #pragma unroll
                for (int i = 0; i < 16; ++i)
                    bm[i] = (__bf16)xb[(size_t)(s0 + g16 + i) * D_ + g * 16 + n];
                cacc[g] = __builtin_amdgcn_wmma_f32_16x16x32_bf16(false, a, false, bm,
                                                                  (short)0, cacc[g], false, false);
            }
        }
        if (!(lane & 16)) {  // rows M=0..7 live here; heads are M<4
            #pragma unroll
            for (int j = 0; j < 4; ++j)
                #pragma unroll
                for (int g = 0; g < 4; ++g)
                    s_part[wave][j][g * 16 + n] = cacc[g][j];
        }
    }
    __syncthreads();
    {
        int h = tid >> 6, e = tid & 63;
        float acc = 0.f;
        #pragma unroll
        for (int w = 0; w < 8; ++w) acc += s_part[w][h][e];
        s_xw[h][e] = acc / s_hsum[h];
    }
    __syncthreads();

    // ---------------- Phase 3: ctx -> value -> hidden/write_value -> logits -> softmax
    if (tid < 64) {
        int h = tid >> 4;
        float acc = bv[tid];
        #pragma unroll 8
        for (int e = 0; e < 64; ++e) acc += wv[tid * 64 + e] * s_xw[h][e];
        s_ctx[tid] = acc;
    }
    __syncthreads();
    if (tid < 64) {
        float acc = bo[tid];
        #pragma unroll 8
        for (int e = 0; e < 64; ++e) acc += wo[tid * 64 + e] * s_ctx[e];
        s_val[tid] = acc;
    }
    __syncthreads();
    if (tid < 128) {
        float acc = b_a1[tid];
        #pragma unroll 8
        for (int d = 0; d < 64; ++d) acc += w_a1[tid * 64 + d] * s_val[d];
        s_hid[tid] = fmaxf(acc, 0.f);
    } else if (tid < 192) {
        int d = tid - 128;
        float acc = b_write[d];
        #pragma unroll 8
        for (int e = 0; e < 64; ++e) acc += w_write[d * 64 + e] * s_val[e];
        s_wvo[d] = acc;
    }
    __syncthreads();
    for (int s = tid; s < S_; s += 256) {
        const float* wr = w_a2 + (size_t)s * 128;
        float acc = b_a2[s];
        #pragma unroll 8
        for (int i = 0; i < 128; ++i) acc += s_hid[i] * wr[i];
        s_lg[s] = acc;
    }
    __syncthreads();
    {
        float mx = -1e30f;
        for (int s = tid; s < S_; s += 256) mx = fmaxf(mx, s_lg[s]);
        s_red[tid] = mx;
        __syncthreads();
        for (int st = 128; st > 0; st >>= 1) {
            if (tid < st) s_red[tid] = fmaxf(s_red[tid], s_red[tid + st]);
            __syncthreads();
        }
        if (tid == 0) s_asc[0] = s_red[0];
        __syncthreads();
        float m = s_asc[0], sum = 0.f;
        for (int s = tid; s < S_; s += 256) {
            float e = __expf(s_lg[s] - m);
            s_lg[s] = e;
            sum += e;
        }
        s_red[tid] = sum;
        __syncthreads();
        for (int st = 128; st > 0; st >>= 1) {
            if (tid < st) s_red[tid] += s_red[tid + st];
            __syncthreads();
        }
        if (tid == 0) s_asc[1] = 1.0f / s_red[0];
    }
    __syncthreads();

    // ---------------- Phase 4: blend out = x*(1-a) + write_value*a  (float4, coalesced)
    {
        const float inv = s_asc[1];
        const float4* __restrict__ x4 = (const float4*)xb;
        float4* __restrict__ o4 = (float4*)(out + (size_t)b * S_ * D_);
        for (int idx = tid; idx < S_ * D_ / 4; idx += 256) {
            int s = idx >> 4, c = idx & 15;
            float a = s_lg[s] * inv;
            float4 xv = x4[idx];
            float4 wvv = *(const float4*)(s_wvo + c * 4);
            float4 r;
            r.x = xv.x * (1.f - a) + wvv.x * a;
            r.y = xv.y * (1.f - a) + wvv.y * a;
            r.z = xv.z * (1.f - a) + wvv.z * a;
            r.w = xv.w * (1.f - a) + wvv.w * a;
            o4[idx] = r;
        }
    }
}

extern "C" void kernel_launch(void* const* d_in, const int* in_sizes, int n_in,
                              void* d_out, int out_size, void* d_ws, size_t ws_size,
                              hipStream_t stream) {
    (void)in_sizes; (void)n_in; (void)d_ws; (void)ws_size; (void)out_size;
    const float* query    = (const float*)d_in[0];
    const float* scr      = (const float*)d_in[1];
    const float* wq       = (const float*)d_in[2];
    const float* wk       = (const float*)d_in[3];
    const float* wv       = (const float*)d_in[4];
    const float* bq       = (const float*)d_in[5];
    const float* bk       = (const float*)d_in[6];
    const float* bv       = (const float*)d_in[7];
    const float* wo       = (const float*)d_in[8];
    const float* bo       = (const float*)d_in[9];
    const float* w_write  = (const float*)d_in[10];
    const float* b_write  = (const float*)d_in[11];
    const float* w_a1     = (const float*)d_in[12];
    const float* b_a1     = (const float*)d_in[13];
    const float* w_a2     = (const float*)d_in[14];
    const float* b_a2     = (const float*)d_in[15];
    float* out = (float*)d_out;

    actformer_fused<<<dim3(B_), dim3(256), 0, stream>>>(
        query, scr, wq, wk, wv, bq, bk, bv, wo, bo,
        w_write, b_write, w_a1, b_a1, w_a2, b_a2, out);
}
